// BranchingQNetwork_27487790694835
// MI455X (gfx1250) — matrix operand order repaired and checked by reading, not compiled, not run on hardware
//
#include <hip/hip_runtime.h>
#include <hip/hip_bf16.h>
#include <stdint.h>

// ---------------------------------------------------------------------------
// BranchingQNetwork on MI455X (gfx1250): bf16 WMMA pipeline.
//   One-time: transpose+convert weights to bf16 N-major (W1 zero-padded K62->64),
//             pack dueling head into 16-column bf16 B matrix.
//   Per 512-row chunk: build px bf16 -> GEMM1(64->2048) -> GEMM2(2048->1024)
//             -> GEMM3(1024->512) -> head WMMA + dueling combine (fp32 out).
// GEMM: 128x128x32 tiles, 8 waves/block, register-staged double-buffered LDS,
//       pure-copy staging (no conversions in the hot loop), global prefetch.
// ---------------------------------------------------------------------------

typedef __attribute__((ext_vector_type(16))) __bf16 v16bf;
typedef __attribute__((ext_vector_type(8)))  float  v8f;

#define NBR    12
#define BATCH  8192
#define OBSD   249
#define HID    512
#define NACT   11
#define CHUNK  512

#define BM 128
#define BN 128
#define BK 32
#define LDSS 40   // ushort stride per LDS row (padded; 80B rows keep 16B alignment)

__device__ __forceinline__ uint16_t f2bf(float f) {
  uint32_t u = __float_as_uint(f);
  uint32_t r = u + 0x7FFFu + ((u >> 16) & 1u);   // round-to-nearest-even
  return (uint16_t)(r >> 16);
}

union Frag16 { uint4 u[2]; v16bf v; };

// ---- build px chunk: [12][CHUNK][64] bf16 (45 shared + 17 group + 2 zero pad)
__global__ void build_px(const float* __restrict__ x, uint16_t* __restrict__ px,
                         int chunkBase) {
  int idx = blockIdx.x * blockDim.x + threadIdx.x;   // 12*CHUNK*64
  int d = idx & 63;
  int r = (idx >> 6) & (CHUNK - 1);
  int i = idx >> 15;                                  // 64*512 = 2^15
  int b = chunkBase + r;
  float v = 0.f;
  if (d < 45)      v = x[(size_t)b * OBSD + d];
  else if (d < 62) v = x[(size_t)b * OBSD + 45 + i * 17 + (d - 45)];
  px[idx] = f2bf(v);
}

// ---- transpose + convert: W (NBR,realK,N) fp32 -> WT (NBR,N,padK) bf16
//      rows realK..padK-1 are zero. 32x32 LDS tiles, coalesced both sides.
__global__ void transpose_w_bf16(const float* __restrict__ W, uint16_t* __restrict__ WT,
                                 int realK, int padK, int N) {
  __shared__ float tile[32][33];
  const int i      = blockIdx.y;
  const int ktiles = padK >> 5;
  const int kBase  = (blockIdx.x % ktiles) * 32;
  const int nBase  = (blockIdx.x / ktiles) * 32;
  const int tr = threadIdx.x >> 5;   // 0..7
  const int tc = threadIdx.x & 31;

  const float* Wb = W + (size_t)i * realK * N;
#pragma unroll
  for (int j = 0; j < 4; ++j) {
    int k = kBase + tr + j * 8;
    tile[tr + j * 8][tc] = (k < realK) ? Wb[(size_t)k * N + nBase + tc] : 0.f;
  }
  __syncthreads();
  uint16_t* WTb = WT + (size_t)i * N * padK;
#pragma unroll
  for (int j = 0; j < 4; ++j) {
    int n = nBase + tr + j * 8;
    WTb[(size_t)n * padK + kBase + tc] = f2bf(tile[tc][tr + j * 8]);
  }
}

// ---- pack head weights: HW[i][n][k] bf16, n<11 = Wa col n, n==11 = Wv, else 0
__global__ void pack_head(const float* __restrict__ Wa, const float* __restrict__ Wv,
                          uint16_t* __restrict__ HW) {
  int idx = blockIdx.x * blockDim.x + threadIdx.x;   // 12*16*512
  int k = idx & 511;
  int n = (idx >> 9) & 15;
  int i = idx >> 13;
  float v = 0.f;
  if (n < NACT)       v = Wa[((size_t)i * HID + k) * NACT + n];
  else if (n == NACT) v = Wv[(size_t)i * HID + k];
  HW[idx] = f2bf(v);
}

// ---- tiled GEMM: Out[br] = relu(A[br] @ WT[br]^T + bias[br]); all matrix bf16
__global__ __launch_bounds__(256)
void gemm_relu_bf16(const uint16_t* __restrict__ A,   // [NBR][CHUNK][K]
                    const uint16_t* __restrict__ WT,  // [NBR][N][K] (N-major)
                    const float*    __restrict__ bias,// [NBR][N]
                    uint16_t*       __restrict__ Out, // [NBR][CHUNK][N]
                    int K, int N, int ntiles) {
  __shared__ __align__(16) uint16_t As[2][BM * LDSS];
  __shared__ __align__(16) uint16_t Bs[2][BN * LDSS];

  const int br    = blockIdx.y;
  const int mBase = (blockIdx.x / ntiles) * BM;
  const int nBase = (blockIdx.x % ntiles) * BN;

  const int tid  = threadIdx.x;
  const int lane = tid & 31;
  const int wave = tid >> 5;
  const int waveM = wave & 3;   // 4 row groups of 32
  const int waveN = wave >> 2;  // 2 col groups of 64
  const int hl  = lane >> 4;    // lane half (K split / M split)
  const int l16 = lane & 15;

  const uint16_t* Ab = A    + (size_t)br * CHUNK * K;
  const uint16_t* Wb = WT   + ((size_t)br * N + nBase) * K;
  const float*    bb = bias + (size_t)br * N;
  uint16_t*       Ob = Out  + (size_t)br * CHUNK * N;

  v8f zero = {};
  v8f acc[2][4];
#pragma unroll
  for (int m = 0; m < 2; ++m)
#pragma unroll
    for (int n = 0; n < 4; ++n) acc[m][n] = zero;

  // staging: thread t -> row t/2 (of 128), two consecutive uint4 (quads {0,1}|{2,3})
  const int sRow  = tid >> 1;
  const int sQuad = (tid & 1) * 2;

  uint4 ra0, ra1, rb0, rb1;   // staged A / B rows (16 bf16 each)

  auto gload = [&](int kBase) {
    const uint16_t* sa = Ab + (size_t)(mBase + sRow) * K + kBase + sQuad * 8;
    ra0 = *(const uint4*)(sa);
    ra1 = *(const uint4*)(sa + 8);
    const uint16_t* sb = Wb + (size_t)sRow * K + kBase + sQuad * 8;
    rb0 = *(const uint4*)(sb);
    rb1 = *(const uint4*)(sb + 8);
  };
  auto lstore = [&](int buf) {
    *(uint4*)&As[buf][sRow * LDSS + sQuad * 8]       = ra0;
    *(uint4*)&As[buf][sRow * LDSS + (sQuad + 1) * 8] = ra1;
    *(uint4*)&Bs[buf][sRow * LDSS + sQuad * 8]       = rb0;
    *(uint4*)&Bs[buf][sRow * LDSS + (sQuad + 1) * 8] = rb1;
  };
  auto pf = [&](int kBase) {      // gfx1250 global_prefetch_b8
    __builtin_prefetch((const void*)(Ab + (size_t)(mBase + sRow) * K + kBase + sQuad * 8), 0, 3);
    __builtin_prefetch((const void*)(Wb + (size_t)sRow * K + kBase + sQuad * 8), 0, 3);
  };

  const int numKB = K / BK;
  gload(0);
  lstore(0);

  for (int kb = 0; kb < numKB; ++kb) {
    __syncthreads();                       // staged tile kb visible to all waves
    const int cur = kb & 1;
    if (kb + 1 < numKB) gload((kb + 1) * BK);   // issue next tile's global loads
    if (kb + 2 < numKB) pf((kb + 2) * BK);

    // A frag: lane l16 = row%16; VGPR0..3 K = hl*8+0..7, VGPR4..7 K = 16+hl*8+0..7
    Frag16 aF[2], bF[4];
#pragma unroll
    for (int m = 0; m < 2; ++m) {
      int row = waveM * 32 + m * 16 + l16;
      const uint16_t* p = &As[cur][row * LDSS + hl * 8];
      aF[m].u[0] = *(const uint4*)(p);
      aF[m].u[1] = *(const uint4*)(p + 16);
    }
    // B frag: lane l16 = col; VGPR0..3 K = hl*8+0..7, VGPR4..7 K = 16+hl*8+0..7
#pragma unroll
    for (int n = 0; n < 4; ++n) {
      int col = waveN * 64 + n * 16 + l16;
      const uint16_t* p = &Bs[cur][col * LDSS + hl * 8];
      bF[n].u[0] = *(const uint4*)(p);
      bF[n].u[1] = *(const uint4*)(p + 16);
    }
#pragma unroll
    for (int m = 0; m < 2; ++m)
#pragma unroll
      for (int n = 0; n < 4; ++n)
        acc[m][n] = __builtin_amdgcn_wmma_f32_16x16x32_bf16(
            false, aF[m].v, false, bF[n].v, (short)0, acc[m][n], false, false);

    if (kb + 1 < numKB) lstore(cur ^ 1);   // stage next tile into alternate buffer
  }

  // epilogue: bias + relu -> bf16. D layout: VGPR r -> M = r + 8*hl, N = l16
#pragma unroll
  for (int n = 0; n < 4; ++n) {
    int col = nBase + waveN * 64 + n * 16 + l16;
    float bcol = bb[col];
#pragma unroll
    for (int m = 0; m < 2; ++m) {
      int rowBase = mBase + waveM * 32 + m * 16 + hl * 8;
#pragma unroll
      for (int r = 0; r < 8; ++r) {
        float v = fmaxf(acc[m][n][r] + bcol, 0.f);
        Ob[(size_t)(rowBase + r) * N + col] = f2bf(v);
      }
    }
  }
}

// ---- head: one wave per 16 rows; h3(16x512) @ HW^T(512x16) then dueling combine
__global__ __launch_bounds__(32)
void head_kernel(const uint16_t* __restrict__ H3,   // [NBR][CHUNK][HID]
                 const uint16_t* __restrict__ HW,   // [NBR][16][HID]
                 const float*    __restrict__ ba,   // [NBR][NACT]
                 const float*    __restrict__ bv,   // [NBR][1]
                 float*          __restrict__ out,  // [NBR][BATCH][NACT]
                 int chunkBase) {
  __shared__ float tile[16][17];
  const int i    = blockIdx.y;
  const int t    = blockIdx.x;
  const int lane = threadIdx.x;
  const int hl   = lane >> 4;
  const int l16  = lane & 15;

  v8f acc = {};
  const uint16_t* Abase = H3 + ((size_t)i * CHUNK + t * 16) * HID;
  const uint16_t* Bbase = HW + (size_t)i * 16 * HID;

#pragma unroll 4
  for (int kb = 0; kb < HID / 32; ++kb) {
    Frag16 aF, bF;
    const uint16_t* pa = Abase + (size_t)l16 * HID + kb * 32 + hl * 8;
    aF.u[0] = *(const uint4*)(pa);
    aF.u[1] = *(const uint4*)(pa + 16);
    const uint16_t* pb = Bbase + (size_t)l16 * HID + kb * 32 + hl * 8;
    bF.u[0] = *(const uint4*)(pb);
    bF.u[1] = *(const uint4*)(pb + 16);
    acc = __builtin_amdgcn_wmma_f32_16x16x32_bf16(
        false, aF.v, false, bF.v, (short)0, acc, false, false);
  }
#pragma unroll
  for (int r = 0; r < 8; ++r) tile[r + hl * 8][l16] = acc[r];
  __syncthreads();

  if (lane < 16) {
    int row = chunkBase + t * 16 + lane;
    float a[NACT], s = 0.f;
#pragma unroll
    for (int j = 0; j < NACT; ++j) { a[j] = tile[lane][j] + ba[i * NACT + j]; s += a[j]; }
    float mean = s * (1.0f / NACT);
    float val  = tile[lane][NACT] + bv[i];
#pragma unroll
    for (int j = 0; j < NACT; ++j)
      out[(size_t)i * BATCH * NACT + (size_t)row * NACT + j] = val + a[j] - mean;
  }
}

// ---------------------------------------------------------------------------
extern "C" void kernel_launch(void* const* d_in, const int* in_sizes, int n_in,
                              void* d_out, int out_size, void* d_ws, size_t ws_size,
                              hipStream_t stream) {
  const float* x  = (const float*)d_in[0];
  const float* W1 = (const float*)d_in[1];
  const float* b1 = (const float*)d_in[2];
  const float* W2 = (const float*)d_in[3];
  const float* b2 = (const float*)d_in[4];
  const float* W3 = (const float*)d_in[5];
  const float* b3 = (const float*)d_in[6];
  const float* Wv = (const float*)d_in[7];
  const float* bv = (const float*)d_in[8];
  const float* Wa = (const float*)d_in[9];
  const float* ba = (const float*)d_in[10];
  float* out = (float*)d_out;
  char*  ws  = (char*)d_ws;

  // workspace layout (all regions naturally 256B aligned)
  const size_t OFF_PX  = 0;                                          // 12*512*64*2
  const size_t OFF_WT1 = OFF_PX  + (size_t)NBR * CHUNK * 64 * 2;     // 12*2048*64*2
  const size_t OFF_WT2 = OFF_WT1 + (size_t)NBR * 2048 * 64 * 2;      // 12*1024*2048*2
  const size_t OFF_WT3 = OFF_WT2 + (size_t)NBR * 1024 * 2048 * 2;    // 12*512*1024*2
  const size_t OFF_H1  = OFF_WT3 + (size_t)NBR * 512 * 1024 * 2;     // 12*512*2048*2
  const size_t OFF_H2  = OFF_H1  + (size_t)NBR * CHUNK * 2048 * 2;   // 12*512*1024*2
  const size_t OFF_H3  = OFF_H2  + (size_t)NBR * CHUNK * 1024 * 2;   // 12*512*512*2
  const size_t OFF_HW  = OFF_H3  + (size_t)NBR * CHUNK * 512 * 2;    // 12*16*512*2

  uint16_t* px  = (uint16_t*)(ws + OFF_PX);
  uint16_t* wt1 = (uint16_t*)(ws + OFF_WT1);
  uint16_t* wt2 = (uint16_t*)(ws + OFF_WT2);
  uint16_t* wt3 = (uint16_t*)(ws + OFF_WT3);
  uint16_t* h1  = (uint16_t*)(ws + OFF_H1);
  uint16_t* h2  = (uint16_t*)(ws + OFF_H2);
  uint16_t* h3  = (uint16_t*)(ws + OFF_H3);
  uint16_t* hw  = (uint16_t*)(ws + OFF_HW);

  // one-time weight prep: bf16, N-major, K padded to multiple of 32
  { dim3 g((64 / 32) * (2048 / 32), NBR);
    transpose_w_bf16<<<g, 256, 0, stream>>>(W1, wt1, 62, 64, 2048); }
  { dim3 g((2048 / 32) * (1024 / 32), NBR);
    transpose_w_bf16<<<g, 256, 0, stream>>>(W2, wt2, 2048, 2048, 1024); }
  { dim3 g((1024 / 32) * (512 / 32), NBR);
    transpose_w_bf16<<<g, 256, 0, stream>>>(W3, wt3, 1024, 1024, 512); }
  pack_head<<<(NBR * 16 * HID) / 256, 256, 0, stream>>>(Wa, Wv, hw);

  for (int c = 0; c < BATCH / CHUNK; ++c) {
    const int base = c * CHUNK;
    build_px<<<(NBR * CHUNK * 64) / 256, 256, 0, stream>>>(x, px, base);

    dim3 g1((CHUNK / BM) * (2048 / BN), NBR);
    gemm_relu_bf16<<<g1, 256, 0, stream>>>(px, wt1, b1, h1, 64, 2048, 2048 / BN);

    dim3 g2((CHUNK / BM) * (1024 / BN), NBR);
    gemm_relu_bf16<<<g2, 256, 0, stream>>>(h1, wt2, b2, h2, 2048, 1024, 1024 / BN);

    dim3 g3((CHUNK / BM) * (512 / BN), NBR);
    gemm_relu_bf16<<<g3, 256, 0, stream>>>(h2, wt3, b3, h3, 1024, 512, 512 / BN);

    dim3 gh(CHUNK / 16, NBR);
    head_kernel<<<gh, 32, 0, stream>>>(h3, hw, ba, bv, out, base);
  }
}
